// Multi_RGCN_54889682042942
// MI455X (gfx1250) — matrix operand order repaired and checked by reading, not compiled, not run on hardware
//
#include <hip/hip_runtime.h>
#include <cstdint>
#include <cstddef>

// ---------------------------------------------------------------------------
// MI455X (gfx1250) implementation: bf16 WMMA GEMMs (v_wmma_f32_16x16x32_bf16)
// for all dense math; L2-resident edge gather + f32 atomic scatter for the
// RGCN message passing; fused LSTM gate pointwise kernel.
// ---------------------------------------------------------------------------

typedef __attribute__((ext_vector_type(16))) __bf16 v16bf;
typedef __attribute__((ext_vector_type(8)))  __bf16 v8bf;
typedef __attribute__((ext_vector_type(8)))  float  v8f;

#define TPB 256

// ---------------- elementwise helpers ----------------
__global__ void fill_zero_f32(float* p, int n) {
    int i = blockIdx.x * blockDim.x + threadIdx.x;
    if (i < n) p[i] = 0.f;
}
__global__ void fill_zero_bf16(__bf16* p, int n) {
    int i = blockIdx.x * blockDim.x + threadIdx.x;
    if (i < n) p[i] = (__bf16)0.f;
}
__global__ void cast_f32_to_bf16(const float* __restrict__ s, __bf16* __restrict__ d, int n) {
    int i = blockIdx.x * blockDim.x + threadIdx.x;
    if (i < n) d[i] = (__bf16)s[i];
}
// dst[n*K + k] = src[k*Nc + n]  (build Bt layout from [K,Nc] f32 weight)
__global__ void transpose_cast(const float* __restrict__ s, __bf16* __restrict__ d, int K, int Nc) {
    int i = blockIdx.x * blockDim.x + threadIdx.x;
    if (i >= K * Nc) return;
    int n = i / K, k = i % K;
    d[i] = (__bf16)s[(size_t)k * Nc + n];
}
// Bt [768][64]: cols 0..703 = convW[r][k][o] (n=r*64+o), cols 704..767 = convS[k][o]
__global__ void prep_rgcn_w(const float* __restrict__ convW, const float* __restrict__ convS,
                            __bf16* __restrict__ d) {
    int i = blockIdx.x * blockDim.x + threadIdx.x;   // 768*64 threads
    if (i >= 768 * 64) return;
    int n = i >> 6, k = i & 63;
    float v;
    if (n < 704) {
        int r = n >> 6, o = n & 63;
        v = convW[(size_t)r * 4096 + (size_t)k * 64 + o];
    } else {
        v = convS[(size_t)k * 64 + (n - 704)];
    }
    d[i] = (__bf16)v;
}

// ---------------- bf16 WMMA GEMM: C[M,Nc] = A[M,K] * Bt[Nc,K]^T ----------------
// A row-major bf16 (lda elems), Bt row-major-in-K bf16 (ldb elems).
// Optional fused bias + relu + bf16 mirror output (same ld as C).
// One wave -> 16x64 output strip; 4 accumulators share one A fragment.
__global__ __launch_bounds__(256)
void gemm_bf16_wmma(const __bf16* __restrict__ A, int lda,
                    const __bf16* __restrict__ Bt, int ldb,
                    float* __restrict__ C, int ldc,
                    int M, int Ncols, int K,
                    const float* __restrict__ bias, int relu,
                    __bf16* __restrict__ Cbf)
{
    const int wave   = blockIdx.x * (blockDim.x >> 5) + (threadIdx.x >> 5);
    const int ntiles = Ncols >> 6;           // 64-wide wave tiles
    const int mtiles = M >> 4;
    if (wave >= mtiles * ntiles) return;
    const int mt = wave / ntiles, nt = wave % ntiles;
    const int lane  = threadIdx.x & 31;
    const int lhalf = lane >> 4;             // 0/1 -> K-half
    const int l15   = lane & 15;
    const int mrow  = (mt << 4) + l15;       // A row for this lane
    const int abase = lhalf << 4;            // A K-offset within chunk (0/16)
    const int bbase = lhalf << 4;            // B K-offset within chunk (0/16)
    const int ncol0 = (nt << 6) + l15;       // first B column for this lane

    v8f acc[4] = {};
    const __bf16* arow = A + (size_t)mrow * lda;
    for (int kk = 0; kk < K; kk += 32) {
        // A fragment: 16 contiguous K values at abase (matches B)
        v8bf alo = *(const v8bf*)(arow + kk + abase);
        v8bf ahi = *(const v8bf*)(arow + kk + abase + 8);
        v16bf af;
        #pragma unroll
        for (int i = 0; i < 8; ++i) { af[i] = alo[i]; af[8 + i] = ahi[i]; }
        #pragma unroll
        for (int t = 0; t < 4; ++t) {
            // B fragment: lane = column, 16 contiguous K values at bbase
            const __bf16* bp = Bt + (size_t)(ncol0 + (t << 4)) * ldb + kk + bbase;
            v16bf bfr = *(const v16bf*)bp;
            acc[t] = __builtin_amdgcn_wmma_f32_16x16x32_bf16(
                false, af, false, bfr, (short)0, acc[t], false, false);
        }
    }
    // C/D layout: element j -> row mt*16 + j + lhalf*8, col = ncol0 + t*16
    const int srow0 = (mt << 4) + (lhalf << 3);
    #pragma unroll
    for (int t = 0; t < 4; ++t) {
        const int n = ncol0 + (t << 4);
        const float bv = bias ? bias[n] : 0.f;
        #pragma unroll
        for (int j = 0; j < 8; ++j) {
            float v = acc[t][j] + bv;
            if (relu) v = fmaxf(v, 0.f);
            size_t off = (size_t)(srow0 + j) * ldc + n;
            C[off] = v;
            if (Cbf) Cbf[off] = (__bf16)v;
        }
    }
}

// ---------------- RGCN edge phase ----------------
// agg[dst[e]][j] += tmp[src[e]][et[e]*64 + j]   (tmp = [N,768], L2-resident)
__global__ void rgcn_scatter(const float* __restrict__ tmp,
                             const int* __restrict__ srcI, const int* __restrict__ dstI,
                             const int* __restrict__ etI, float* __restrict__ agg, int E)
{
    int idx = blockIdx.x * blockDim.x + threadIdx.x;
    int e = idx >> 6, j = idx & 63;
    if (e >= E) return;
    float v = tmp[(size_t)srcI[e] * 768 + (size_t)etI[e] * 64 + j];
    unsafeAtomicAdd(&agg[(size_t)dstI[e] * 64 + j], v);     // global_atomic_add_f32
}

// h_out = relu(agg + tmp[:,704:768] + b); write f32 into o[:,layer*64..] and bf16 next-layer input
__global__ void rgcn_finish(const float* __restrict__ agg, const float* __restrict__ tmp,
                            const float* __restrict__ bias, float* __restrict__ o,
                            __bf16* __restrict__ hnext, int layer, int N)
{
    int idx = blockIdx.x * blockDim.x + threadIdx.x;
    int n = idx >> 6, j = idx & 63;
    if (n >= N) return;
    float v = agg[(size_t)n * 64 + j] + tmp[(size_t)n * 768 + 704 + j] + bias[j];
    v = fmaxf(v, 0.f);
    o[(size_t)n * 128 + layer * 64 + j] = v;
    hnext[(size_t)n * 64 + j] = (__bf16)v;
}

// em[t][b][0:128]=o[enroll[b]], [128:256]=o[course[b]]  (bf16, LSTM A input)
__global__ void gather_em(const float* __restrict__ o, const int* __restrict__ enr,
                          const int* __restrict__ crs, __bf16* __restrict__ em, int t)
{
    int idx = blockIdx.x * blockDim.x + threadIdx.x;   // 2048*256
    int b = idx >> 8, j = idx & 255;
    if (b >= 2048) return;
    int row = (j < 128) ? enr[b] : crs[b];
    em[((size_t)t * 2048 + b) * 256 + j] = (__bf16)o[(size_t)row * 128 + (j & 127)];
}

// ---------------- LSTM fused gate/state kernel ----------------
__global__ void lstm_cell(const float* __restrict__ gx, const float* __restrict__ gh,
                          const float* __restrict__ bih, const float* __restrict__ bhh,
                          float* __restrict__ c, __bf16* __restrict__ hbf,
                          __bf16* __restrict__ xout, int xld, int coloff, int Bq)
{
    int idx = blockIdx.x * blockDim.x + threadIdx.x;   // Bq*256
    int b = idx >> 8, j = idx & 255;
    if (b >= Bq) return;
    const float* gxr = gx + (size_t)b * 1024;
    const float* ghr = gh + (size_t)b * 1024;
    float gi = gxr[j]       + ghr[j]       + bih[j]       + bhh[j];
    float gf = gxr[256 + j] + ghr[256 + j] + bih[256 + j] + bhh[256 + j];
    float gg = gxr[512 + j] + ghr[512 + j] + bih[512 + j] + bhh[512 + j];
    float go = gxr[768 + j] + ghr[768 + j] + bih[768 + j] + bhh[768 + j];
    gi = 1.f / (1.f + __expf(-gi));
    gf = 1.f / (1.f + __expf(-gf));
    go = 1.f / (1.f + __expf(-go));
    gg = tanhf(gg);
    float cn = gf * c[(size_t)b * 256 + j] + gi * gg;
    c[(size_t)b * 256 + j] = cn;
    float h = go * tanhf(cn);
    hbf[(size_t)b * 256 + j] = (__bf16)h;
    xout[(size_t)b * xld + coloff + j] = (__bf16)h;
}

// ---------------- head: out[b] = sigmoid(x2[b] . W3 + b3) ----------------
__global__ void head_final(const float* __restrict__ x2, const float* __restrict__ W3,
                           const float* __restrict__ b3, float* __restrict__ out, int Bq)
{
    int b = blockIdx.x * blockDim.x + threadIdx.x;
    if (b >= Bq) return;
    float s = b3[0];
    #pragma unroll 4
    for (int j = 0; j < 128; ++j) s += x2[(size_t)b * 128 + j] * W3[j];
    out[b] = 1.f / (1.f + __expf(-s));
}

// ---------------------------------------------------------------------------
extern "C" void kernel_launch(void* const* d_in, const int* in_sizes, int n_in,
                              void* d_out, int out_size, void* d_ws, size_t ws_size,
                              hipStream_t stream)
{
    const int N  = in_sizes[0] / 64;   // 30000 (multiple of 16)
    const int E  = in_sizes[1];        // 480000
    const int Bq = 2048, T = 3;

    // ---- params (alphabetical flatten after 19 positional inputs) ----
    const int P = 19;
    const float* W1p = (const float*)d_in[P + 0];
    const float* W2p = (const float*)d_in[P + 1];
    const float* W3p = (const float*)d_in[P + 2];
    const float* Whh[2][2]; const float* Wih[2][2];
    const float* bhh[2][2]; const float* bih[2][2];
    for (int l = 0; l < 2; ++l)
        for (int d = 0; d < 2; ++d) {
            Whh[l][d] = (const float*)d_in[P + 3  + l * 2 + d];
            Wih[l][d] = (const float*)d_in[P + 7  + l * 2 + d];
            bhh[l][d] = (const float*)d_in[P + 14 + l * 2 + d];
            bih[l][d] = (const float*)d_in[P + 18 + l * 2 + d];
        }
    const float* b1p = (const float*)d_in[P + 11];
    const float* b2p = (const float*)d_in[P + 12];
    const float* b3p = (const float*)d_in[P + 13];
    const float* convB[2] = { (const float*)d_in[P + 22], (const float*)d_in[P + 23] };
    const float* convS[2] = { (const float*)d_in[P + 24], (const float*)d_in[P + 25] };
    const float* convW[2] = { (const float*)d_in[P + 26], (const float*)d_in[P + 27] };

    // ---- workspace carve (256B aligned; WMMA loads need 32B alignment) ----
    char* wp = (char*)d_ws;
    auto carve = [&](size_t bytes) -> char* {
        char* r = wp; wp += (bytes + 255) & ~(size_t)255; return r;
    };
    __bf16* wih_bf[2][2]; __bf16* whh_bf[2][2];
    for (int l = 0; l < 2; ++l)
        for (int d = 0; d < 2; ++d) {
            wih_bf[l][d] = (__bf16*)carve((size_t)1024 * (l == 0 ? 256 : 512) * 2);
            whh_bf[l][d] = (__bf16*)carve((size_t)1024 * 256 * 2);
        }
    __bf16* w1t   = (__bf16*)carve((size_t)256 * 512 * 2);
    __bf16* w2t   = (__bf16*)carve((size_t)128 * 256 * 2);
    __bf16* wr[2] = { (__bf16*)carve((size_t)768 * 64 * 2), (__bf16*)carve((size_t)768 * 64 * 2) };
    __bf16* h_bf  = (__bf16*)carve((size_t)N * 64 * 2);
    float*  tmp   = (float*) carve((size_t)N * 768 * 4);   // [N,768] all-relation transform (fits L2)
    float*  agg   = (float*) carve((size_t)N * 64 * 4);
    float*  obuf  = (float*) carve((size_t)N * 128 * 4);
    __bf16* em_bf = (__bf16*)carve((size_t)T * Bq * 256 * 2);
    float*  gx    = (float*) carve((size_t)T * Bq * 1024 * 4);
    float*  ghb   = (float*) carve((size_t)Bq * 1024 * 4);
    __bf16* hbf   = (__bf16*)carve((size_t)Bq * 256 * 2);
    float*  cst   = (float*) carve((size_t)Bq * 256 * 4);
    __bf16* xbuf0 = (__bf16*)carve((size_t)T * Bq * 512 * 2);
    __bf16* xbuf1 = (__bf16*)carve((size_t)T * Bq * 512 * 2);
    float*  hx1f  = (float*) carve((size_t)Bq * 256 * 4);
    __bf16* hx1b  = (__bf16*)carve((size_t)Bq * 256 * 2);
    float*  hx2f  = (float*) carve((size_t)Bq * 128 * 4);
    __bf16* hx2b  = (__bf16*)carve((size_t)Bq * 128 * 2);

    auto blocks1d = [](size_t n) { return (unsigned)((n + TPB - 1) / TPB); };
    auto castb = [&](const float* s, __bf16* d, int n) {
        cast_f32_to_bf16<<<blocks1d(n), TPB, 0, stream>>>(s, d, n);
    };
    auto fillf = [&](float* ptr, int n) {
        fill_zero_f32<<<blocks1d(n), TPB, 0, stream>>>(ptr, n);
    };
    auto fillb = [&](__bf16* ptr, int n) {
        fill_zero_bf16<<<blocks1d(n), TPB, 0, stream>>>(ptr, n);
    };
    auto gemm = [&](const __bf16* A, int lda, const __bf16* Bt, int ldb,
                    float* C, int ldc, int M, int Nc, int K,
                    const float* bias, int relu, __bf16* Cbf) {
        int waves = (M >> 4) * (Nc >> 6);
        gemm_bf16_wmma<<<(waves + 7) / 8, 256, 0, stream>>>(
            A, lda, Bt, ldb, C, ldc, M, Nc, K, bias, relu, Cbf);
    };

    // ---- weight prep (f32 -> bf16, Bt layout) ----
    for (int l = 0; l < 2; ++l)
        for (int d = 0; d < 2; ++d) {
            castb(Wih[l][d], wih_bf[l][d], 1024 * (l == 0 ? 256 : 512));
            castb(Whh[l][d], whh_bf[l][d], 1024 * 256);
        }
    transpose_cast<<<blocks1d(512 * 256), TPB, 0, stream>>>(W1p, w1t, 512, 256);
    transpose_cast<<<blocks1d(256 * 128), TPB, 0, stream>>>(W2p, w2t, 256, 128);
    for (int l = 0; l < 2; ++l)
        prep_rgcn_w<<<blocks1d(768 * 64), TPB, 0, stream>>>(convW[l], convS[l], wr[l]);

    // ---- 3 graphs: RGCN (GEMM + scatter) + embedding gather ----
    for (int g = 0; g < 3; ++g) {
        const float* hin  = (const float*)d_in[g * 6 + 0];
        const int*   srcI = (const int*)  d_in[g * 6 + 1];
        const int*   dstI = (const int*)  d_in[g * 6 + 2];
        const int*   etI  = (const int*)  d_in[g * 6 + 3];
        const int*   enr  = (const int*)  d_in[g * 6 + 4];
        const int*   crs  = (const int*)  d_in[g * 6 + 5];
        castb(hin, h_bf, N * 64);
        for (int l = 0; l < 2; ++l) {
            // all-relation + self-loop transform in one [N,64]x[64,768] WMMA GEMM
            gemm(h_bf, 64, wr[l], 64, tmp, 768, N, 768, 64, nullptr, 0, nullptr);
            fillf(agg, N * 64);
            rgcn_scatter<<<blocks1d((size_t)E * 64), TPB, 0, stream>>>(tmp, srcI, dstI, etI, agg, E);
            rgcn_finish<<<blocks1d((size_t)N * 64), TPB, 0, stream>>>(agg, tmp, convB[l], obuf, h_bf, l, N);
        }
        gather_em<<<blocks1d((size_t)Bq * 256), TPB, 0, stream>>>(obuf, enr, crs, em_bf, g);
    }

    // ---- 2-layer bidirectional LSTM (T=3) ----
    for (int l = 0; l < 2; ++l) {
        int Din = (l == 0) ? 256 : 512;
        const __bf16* X  = (l == 0) ? em_bf : xbuf0;
        __bf16*       XO = (l == 0) ? xbuf0 : xbuf1;
        for (int d = 0; d < 2; ++d) {
            // all timesteps at once: [T*B, Din] x [Din, 1024]
            gemm(X, Din, wih_bf[l][d], Din, gx, 1024, T * Bq, 1024, Din, nullptr, 0, nullptr);
            fillb(hbf, Bq * 256);
            fillf(cst, Bq * 256);
            for (int s = 0; s < T; ++s) {
                int t = (d == 0) ? s : (T - 1 - s);
                gemm(hbf, 256, whh_bf[l][d], 256, ghb, 1024, Bq, 1024, 256, nullptr, 0, nullptr);
                lstm_cell<<<blocks1d((size_t)Bq * 256), TPB, 0, stream>>>(
                    gx + (size_t)t * Bq * 1024, ghb, bih[l][d], bhh[l][d],
                    cst, hbf, XO + (size_t)t * Bq * 512, 512, d * 256, Bq);
            }
        }
    }

    // ---- head: relu(last@W1+b1) -> relu(@W2+b2) -> sigmoid(@W3+b3) ----
    const __bf16* last = xbuf1 + (size_t)(T - 1) * Bq * 512;
    gemm(last, 512, w1t, 512, hx1f, 256, Bq, 256, 512, b1p, 1, hx1b);
    gemm(hx1b, 256, w2t, 256, hx2f, 128, Bq, 128, 256, b2p, 1, hx2b);
    head_final<<<blocks1d(Bq), TPB, 0, stream>>>(hx2f, W3p, b3p, (float*)d_out, Bq);

    (void)n_in; (void)out_size; (void)ws_size;
}
